// SiamesePredictionNet_24034636989217
// MI455X (gfx1250) — matrix operand order repaired and checked by reading, not compiled, not run on hardware
//
#include <hip/hip_runtime.h>

// ---------------------------------------------------------------------------
// Types
// ---------------------------------------------------------------------------
typedef __bf16 bf16;
typedef bf16     bf16x8  __attribute__((ext_vector_type(8)));
typedef bf16     bf16x16 __attribute__((ext_vector_type(16)));
typedef float    f32x8   __attribute__((ext_vector_type(8)));
typedef unsigned v4u     __attribute__((ext_vector_type(4)));
typedef int      v4i     __attribute__((ext_vector_type(4)));
typedef int      v8i     __attribute__((ext_vector_type(8)));

#define SEQ   (128 * 256)   // 32768 total timesteps
#define IN    64
#define HD    128
#define GD    512           // 4*H gates
#define LLAST 256
#define WPAD  136           // padded bf16 row length for LDS Whh (kills bank conflicts)

#if defined(__has_builtin)
#if __has_builtin(__builtin_amdgcn_tensor_load_to_lds)
#define HAVE_TDM 1
#endif
#if __has_builtin(__builtin_amdgcn_s_wait_tensorcnt)
#define HAVE_WAIT_TENSOR 1
#endif
#endif

__device__ inline void wait_tensorcnt0() {
#if defined(HAVE_WAIT_TENSOR)
    __builtin_amdgcn_s_wait_tensorcnt(0);
#else
    asm volatile("s_wait_tensorcnt 0x0" ::: "memory");
#endif
}

// ---------------------------------------------------------------------------
// f32 -> bf16 conversion (elementwise)
// ---------------------------------------------------------------------------
__global__ void cvt_f32_bf16(const float* __restrict__ in, bf16* __restrict__ out, int n) {
    int i = blockIdx.x * blockDim.x + threadIdx.x;
    if (i < n) out[i] = (bf16)in[i];
}

__global__ void add_vec(const float* __restrict__ a, const float* __restrict__ b,
                        float* __restrict__ o, int n) {
    int i = blockIdx.x * blockDim.x + threadIdx.x;
    if (i < n) o[i] = a[i] + b[i];
}

// ---------------------------------------------------------------------------
// WMMA fragment loaders (bf16, 16x16x32), per CDNA5 ISA 7.12.2 layouts.
// A (16Mx32K): lane<16 -> M=lane, elems 0..7=K0..7, 8..15=K16..23
//              lane>=16 -> M=lane-16, elems 0..7=K8..15, 8..15=K24..31
// B (32Kx16N): lane<16 -> N=lane, elems 0..7=K0..7, 8..15=K16..23
//              lane>=16 -> N=lane-16, elems 0..7=K8..15, 8..15=K24..31
// ---------------------------------------------------------------------------
__device__ inline bf16x16 load_a_frag_lds(const bf16* sm, int row_halves,
                                          int rowbase, int lane, int k0) {
    int row = rowbase + (lane & 15);
    int kb  = k0 + ((lane >> 4) << 3);
    const bf16* p = sm + row * row_halves + kb;
    bf16x8 lo = *reinterpret_cast<const bf16x8*>(p);
    bf16x8 hi = *reinterpret_cast<const bf16x8*>(p + 16);
    bf16x16 r;
#pragma unroll
    for (int i = 0; i < 8; ++i) { r[i] = lo[i]; r[i + 8] = hi[i]; }
    return r;
}

__device__ inline bf16x16 load_b_frag(const bf16* __restrict__ tile, int ld, int lane, int k0) {
    int col = lane & 15;
    int kb  = k0 + ((lane >> 4) << 3);
    const bf16* p = tile + (long)col * ld + kb;
    bf16x8 lo = *reinterpret_cast<const bf16x8*>(p);
    bf16x8 hi = *reinterpret_cast<const bf16x8*>(p + 16);
    bf16x16 r;
#pragma unroll
    for (int i = 0; i < 8; ++i) { r[i] = lo[i]; r[i + 8] = hi[i]; }
    return r;
}

// ---------------------------------------------------------------------------
// GEMM: C[m,n] = act(A[m,:] . B[n,:] + bias[n])   (A @ B^T, B row-major [N,K])
// Block = 256 threads (8 waves) computing a 64M x 128N tile.
//   - A tile (64 x KDIM bf16) staged to LDS once per block by the Tensor Data
//     Mover (one tensor_load_to_lds issued by wave 0; TENSORcnt wait; barrier).
//     TDM inserts 4 dwords of padding per row -> 16B-aligned, bank-spread rows.
//   - Each wave owns a 16-wide N strip: per K-step, the B fragment and all 4 A
//     fragments are preloaded, then 4 independent WMMAs issue back-to-back.
// grid = (M/64, N/128, 2 branches)
// ---------------------------------------------------------------------------
template <int KDIM, bool RELU, bool OUT_BF16>
__global__ __launch_bounds__(256) void wmma_gemm_bias(
    const bf16* __restrict__ A, long strideA,     // per-branch elem stride
    const bf16* __restrict__ B,                   // shared weights [N,KDIM]
    const float* __restrict__ bias,               // [N]
    float* __restrict__ Cf, long strideCf,        // f32 out (or null)
    bf16* __restrict__ Cb, long strideCb,         // bf16 out (or null)
    int N) {
    constexpr int ROW_DW   = KDIM / 2;            // dwords per unpadded row
    constexpr int ROW_DW_P = ROW_DW + 4;          // padded stride in dwords
    constexpr int ROW_H_P  = KDIM + 8;            // padded stride in halves
    __shared__ bf16 smA[64 * ROW_H_P];

    const int lane = threadIdx.x & 31;
    const int wv   = threadIdx.x >> 5;
    const int m0   = blockIdx.x * 64;
    const int n0   = blockIdx.y * 128 + wv * 16;
    const int br   = blockIdx.z;

    const bf16* Abase = A + (long)br * strideA + (long)m0 * KDIM;
    const bf16* Bbase = B + (long)n0 * KDIM;

#if defined(HAVE_TDM)
    if (wv == 0) {
        // Tensor DMA descriptor (CDNA5 ISA ch.8): 2D tile, bf16 elements,
        // pad 4 dwords after every row of ROW_DW dwords.
        constexpr unsigned PAD_ITV = (ROW_DW == 32) ? 4u : 5u;  // 2^(i+1) dwords
        unsigned lds_base = (unsigned)(uintptr_t)&smA[0];
        unsigned long long ga = (unsigned long long)(uintptr_t)Abase;
        v4u g0;
        g0[0] = 1u;                                   // count=1 (valid), user mode
        g0[1] = lds_base;                             // lds_addr
        g0[2] = (unsigned)(ga & 0xFFFFFFFFull);       // global_addr[31:0]
        g0[3] = (unsigned)((ga >> 32) & 0x1FFFFFFull) // global_addr[56:32]
              | (2u << 30);                           // type = 2 (image)
        v8i g1;
        g1[0] = (int)((1u << 16)                      // data_size = 2B
              | (1u << 20)                            // pad_enable
              | (PAD_ITV << 22)                       // pad_interval
              | (3u << 25));                          // pad_amount = 4 dwords
        g1[1] = (int)((unsigned)(KDIM & 0xFFFF) << 16);      // tensor_dim0 lo16
        g1[2] = (int)((KDIM >> 16) | (64u << 16));           // dim0 hi | dim1 lo
        g1[3] = (int)((unsigned)KDIM << 16);                 // dim1 hi | tile_dim0
        g1[4] = 64;                                          // tile_dim1=64, tile_dim2=0
        g1[5] = KDIM;                                        // tensor_dim0_stride lo32
        g1[6] = 0;
        g1[7] = 0;
        v4i g2 = {0, 0, 0, 0};
        v4i g3 = {0, 0, 0, 0};
#if __clang_major__ >= 23
        v8i g4 = {0, 0, 0, 0, 0, 0, 0, 0};
        __builtin_amdgcn_tensor_load_to_lds(g0, g1, g2, g3, g4, 0);
#else
        __builtin_amdgcn_tensor_load_to_lds(g0, g1, g2, g3, 0);
#endif
        wait_tensorcnt0();
    }
#else
    // Fallback: cooperative staged copy with the same padded layout.
    for (int idx = threadIdx.x; idx < 64 * ROW_DW; idx += 256) {
        int r = idx / ROW_DW, d = idx % ROW_DW;
        ((unsigned*)smA)[r * ROW_DW_P + d] = ((const unsigned*)Abase)[r * ROW_DW + d];
    }
#endif
    __syncthreads();

    f32x8 acc[4] = {};
#pragma unroll
    for (int k0 = 0; k0 < KDIM; k0 += 32) {
        // Preload B fragment and all 4 A fragments, then issue the 4
        // independent WMMAs back-to-back (waits amortized, co-execution).
        bf16x16 b = load_b_frag(Bbase, KDIM, lane, k0);
        bf16x16 a0 = load_a_frag_lds(smA, ROW_H_P,  0, lane, k0);
        bf16x16 a1 = load_a_frag_lds(smA, ROW_H_P, 16, lane, k0);
        bf16x16 a2 = load_a_frag_lds(smA, ROW_H_P, 32, lane, k0);
        bf16x16 a3 = load_a_frag_lds(smA, ROW_H_P, 48, lane, k0);
        acc[0] = __builtin_amdgcn_wmma_f32_16x16x32_bf16(
                     false, a0, false, b, (short)0, acc[0], false, false);
        acc[1] = __builtin_amdgcn_wmma_f32_16x16x32_bf16(
                     false, a1, false, b, (short)0, acc[1], false, false);
        acc[2] = __builtin_amdgcn_wmma_f32_16x16x32_bf16(
                     false, a2, false, b, (short)0, acc[2], false, false);
        acc[3] = __builtin_amdgcn_wmma_f32_16x16x32_bf16(
                     false, a3, false, b, (short)0, acc[3], false, false);
    }

    // Epilogue: 32-bit flat indexing (max index S*GD = 16.7M << 2^31) keeps
    // the address math out of the 64-bit VALU path.
    const int col  = n0 + (lane & 15);
    const float bv = bias[col];
    float* __restrict__ Cfp = Cf + (long)br * strideCf;
    bf16*  __restrict__ Cbp = Cb + (long)br * strideCb;
#pragma unroll
    for (int mt = 0; mt < 4; ++mt) {
        const unsigned base =
            (unsigned)(m0 + mt * 16 + ((lane >> 4) << 3)) * (unsigned)N + (unsigned)col;
#pragma unroll
        for (int r = 0; r < 8; ++r) {
            float v = acc[mt][r] + bv;
            if (RELU) v = v > 0.f ? v : 0.f;
            unsigned idx = base + (unsigned)r * (unsigned)N;
            if (OUT_BF16) Cbp[idx] = (bf16)v;
            else          Cfp[idx] = v;
        }
    }
}

// ---------------------------------------------------------------------------
// LSTM recurrence for one layer. grid.x = branch (2 blocks), 512 threads.
// Whh kept resident in LDS as bf16 with padded rows; h,c,gates in LDS.
// Gate order (PyTorch): i, f, g, o. This is the sequential critical path:
// 32768 dependent steps of a 128->512 matvec + nonlinearities.
// ---------------------------------------------------------------------------
__global__ __launch_bounds__(512) void lstm_recurrence(
    const float* __restrict__ xg, long xg_stride,       // per-branch [S][512]
    const bf16* __restrict__ Whh,                       // [512][128] bf16
    bf16* __restrict__ h_hist, long h_stride,           // [S][128] bf16 or null
    float* __restrict__ y_last, long y_stride,          // [LLAST][128] or null
    int S) {
    __shared__ bf16  w[GD * WPAD];
    __shared__ float hs[HD];
    __shared__ float cs[HD];
    __shared__ float g[GD];

    const int j  = threadIdx.x;
    const int br = blockIdx.x;
    xg += (long)br * xg_stride;

    // Stage recurrent weights into LDS (row j -> padded row j)
#pragma unroll
    for (int k = 0; k < HD; k += 8) {
        *reinterpret_cast<bf16x8*>(&w[j * WPAD + k]) =
            *reinterpret_cast<const bf16x8*>(&Whh[(long)j * HD + k]);
    }
    if (j < HD) { hs[j] = 0.f; cs[j] = 0.f; }
    __syncthreads();

    const bf16* wr = &w[j * WPAD];
    for (int t = 0; t < S; ++t) {
        if (t + 8 < S)
            __builtin_prefetch(&xg[(long)(t + 8) * GD + j], 0, 1);

        float acc = xg[(long)t * GD + j];
#pragma unroll
        for (int k = 0; k < HD; k += 8) {
            bf16x8 wv = *reinterpret_cast<const bf16x8*>(wr + k);
#pragma unroll
            for (int u = 0; u < 8; ++u) acc += (float)wv[u] * hs[k + u];
        }
        float a = ((j >> 7) == 2) ? tanhf(acc) : (1.f / (1.f + __expf(-acc)));
        g[j] = a;
        __syncthreads();   // all hs reads + gate writes done

        if (j < HD) {
            float c = g[HD + j] * cs[j] + g[j] * g[2 * HD + j];
            cs[j] = c;
            float h = g[3 * HD + j] * tanhf(c);
            hs[j] = h;
            if (h_hist) h_hist[(long)br * h_stride + (long)t * HD + j] = (bf16)h;
            if (y_last && t >= S - LLAST)
                y_last[(long)br * y_stride + (long)(t - (S - LLAST)) * HD + j] = h;
        }
        __syncthreads();   // hs update visible before next step's dot
    }
}

// ---------------------------------------------------------------------------
// Head: p = W_head . relu(+/-(y1-y2)) + b, softmax over [p1,p2,pd]
// ---------------------------------------------------------------------------
__global__ void head_softmax(const float* __restrict__ y,   // [2][LLAST][HD]
                             const float* __restrict__ Wh,  // [HD]
                             const float* __restrict__ bh,  // [1]
                             float* __restrict__ out, int L) {
    int l = blockIdx.x * blockDim.x + threadIdx.x;
    if (l >= L) return;
    const float* y1 = y + (long)l * HD;
    const float* y2 = y + (long)L * HD + (long)l * HD;
    float p1 = 0.f, p2 = 0.f, pd = 0.f;
#pragma unroll 4
    for (int k = 0; k < HD; ++k) {
        float d   = y1[k] - y2[k];
        float r12 = d > 0.f ? d : 0.f;
        float r21 = d < 0.f ? -d : 0.f;
        float wk  = Wh[k];
        p1 += wk * r12;
        p2 += wk * r21;
        pd += wk * (r12 - r21);
    }
    float b = bh[0];
    p1 += b; p2 += b; pd += b;
    float m  = fmaxf(p1, fmaxf(p2, pd));
    float e1 = __expf(p1 - m), e2 = __expf(p2 - m), e3 = __expf(pd - m);
    float s  = e1 + e2 + e3;
    out[l * 3 + 0] = e1 / s;
    out[l * 3 + 1] = e2 / s;
    out[l * 3 + 2] = e3 / s;
}

// ---------------------------------------------------------------------------
// Host orchestration
// ---------------------------------------------------------------------------
extern "C" void kernel_launch(void* const* d_in, const int* in_sizes, int n_in,
                              void* d_out, int out_size, void* d_ws, size_t ws_size,
                              hipStream_t stream) {
    const float* inp1   = (const float*)d_in[0];  // [S][64]
    const float* inp2   = (const float*)d_in[1];
    const float* W_fc   = (const float*)d_in[2];  // [128][64]
    const float* b_fc   = (const float*)d_in[3];  // [128]
    const float* W_ih   = (const float*)d_in[4];  // [2][512][128]
    const float* W_hh   = (const float*)d_in[5];  // [2][512][128]
    const float* b_ih   = (const float*)d_in[6];  // [2][512]
    const float* b_hh   = (const float*)d_in[7];  // [2][512]
    const float* W_head = (const float*)d_in[8];  // [128]
    const float* b_head = (const float*)d_in[9];  // [1]
    float* out = (float*)d_out;                   // [256][3]

    const int S = SEQ;

    // ---- workspace layout (all offsets 256B-aligned) -----------------------
    char*  ws  = (char*)d_ws;
    size_t off = 0;
    auto alloc = [&](size_t bytes) { char* p = ws + off; off = (off + bytes + 255) & ~(size_t)255; return p; };

    bf16*  inA  = (bf16*)alloc((size_t)2 * S * IN * sizeof(bf16));      // bf16 inputs, both branches
    bf16*  wfc  = (bf16*)alloc((size_t)HD * IN * sizeof(bf16));
    bf16*  wih  = (bf16*)alloc((size_t)2 * GD * HD * sizeof(bf16));
    bf16*  whh  = (bf16*)alloc((size_t)2 * GD * HD * sizeof(bf16));
    float* bc   = (float*)alloc((size_t)2 * GD * sizeof(float));        // b_ih + b_hh
    bf16*  xfc  = (bf16*)alloc((size_t)2 * S * HD * sizeof(bf16));      // fc output
    bf16*  h1   = (bf16*)alloc((size_t)2 * S * HD * sizeof(bf16));      // layer-1 hidden history
    float* yv   = (float*)alloc((size_t)2 * LLAST * HD * sizeof(float));
    float* xg   = (float*)alloc((size_t)2 * S * GD * sizeof(float));    // gate preactivations (reused)

    // ---- conversions -------------------------------------------------------
    auto cvt = [&](const float* src, bf16* dst, int n) {
        cvt_f32_bf16<<<(n + 255) / 256, 256, 0, stream>>>(src, dst, n);
    };
    cvt(inp1, inA, S * IN);
    cvt(inp2, inA + (size_t)S * IN, S * IN);
    cvt(W_fc, wfc, HD * IN);
    cvt(W_ih, wih, 2 * GD * HD);
    cvt(W_hh, whh, 2 * GD * HD);
    add_vec<<<(2 * GD + 255) / 256, 256, 0, stream>>>(b_ih, b_hh, bc, 2 * GD);

    // ---- FC: relu(inp @ W_fc^T + b_fc) -> xfc (bf16) -----------------------
    wmma_gemm_bias<IN, true, true><<<dim3(S / 64, HD / 128, 2), 256, 0, stream>>>(
        inA, (long)S * IN, wfc, b_fc,
        (float*)nullptr, 0, xfc, (long)S * HD, HD);

    // ---- layer 1: xg = xfc @ Wih0^T + (bih0+bhh0) --------------------------
    wmma_gemm_bias<HD, false, false><<<dim3(S / 64, GD / 128, 2), 256, 0, stream>>>(
        xfc, (long)S * HD, wih, bc,
        xg, (long)S * GD, (bf16*)nullptr, 0, GD);

    lstm_recurrence<<<dim3(2), 512, 0, stream>>>(
        xg, (long)S * GD, whh, h1, (long)S * HD, (float*)nullptr, 0, S);

    // ---- layer 2: xg = h1 @ Wih1^T + (bih1+bhh1) ---------------------------
    wmma_gemm_bias<HD, false, false><<<dim3(S / 64, GD / 128, 2), 256, 0, stream>>>(
        h1, (long)S * HD, wih + (size_t)GD * HD, bc + GD,
        xg, (long)S * GD, (bf16*)nullptr, 0, GD);

    lstm_recurrence<<<dim3(2), 512, 0, stream>>>(
        xg, (long)S * GD, whh + (size_t)GD * HD,
        (bf16*)nullptr, 0, yv, (long)LLAST * HD, S);

    // ---- head + softmax ----------------------------------------------------
    head_softmax<<<(LLAST + 127) / 128, 128, 0, stream>>>(yv, W_head, b_head, out, LLAST);
}
